// SelfRNN_46488726012430
// MI455X (gfx1250) — matrix operand order repaired and checked by reading, not compiled, not run on hardware
//
#include <hip/hip_runtime.h>
#include <hip/hip_bf16.h>

// ---------------------------------------------------------------------------
// 4-layer tanh RNN for MI455X (gfx1250), persistent pipelined WMMA kernel.
//   S=256, B=64, H=512, L=4
//   grid = L * WPL workgroups; each WG owns a 64-column slice of one layer.
//   Weights converted f32->bf16 into LDS once.
//   Producers publish bf16 activations into a ring buffer in d_ws; consumers
//   stage them with global_load_async_to_lds_b128 (ASYNCcnt path, GV mode).
//   Cross-WG ordering via per-(t,layer) atomic counters in d_ws.
// ---------------------------------------------------------------------------

#define SEQ     256
#define BATCH   64
#define HDIM    512
#define NLAYER  4
#define WPL     8            // workgroups per layer (column slices)
#define SLICE   64           // output columns per workgroup
#define STRIDE  528          // padded LDS row stride (bf16 elements)
#define RING    8            // ring depth (timesteps) for bf16 activations

#define REGION_BYTES (64 * STRIDE * 2)      // 67584 B per LDS region
#define A_X_OFF  (0 * REGION_BYTES)
#define A_H_OFF  (1 * REGION_BYTES)
#define LDS_BYTES (4 * REGION_BYTES)        // 270336 B < 320 KB WGP budget

#define FLAGS_BYTES 4096                    // flags region at front of d_ws

typedef __bf16 bf16;
typedef __attribute__((ext_vector_type(16))) __bf16 v16bf;
typedef __attribute__((ext_vector_type(8)))  __bf16 v8bf;
typedef __attribute__((ext_vector_type(8)))  float  v8f;

#if __has_builtin(__builtin_amdgcn_tanhf)
#define fast_tanh(x) __builtin_amdgcn_tanhf(x)
#elif __has_builtin(__builtin_amdgcn_tanh_f32)
#define fast_tanh(x) __builtin_amdgcn_tanh_f32(x)
#else
#define fast_tanh(x) tanhf(x)
#endif

// A-matrix fragment (16x32 bf16, MxK) per ISA 7.12.2:
//   lane = (khalf<<4)|m ; regs 0-3 hold k = khalf*8 + 0..7, regs 4-7 hold k+16
__device__ __forceinline__ v16bf load_a_frag(const bf16* A, int mtile, int kb, int lane) {
    const int m  = mtile * 16 + (lane & 15);
    const int k8 = (lane >> 4) * 8;
    const bf16* p = A + m * STRIDE + kb;
    v8bf lo = *(const v8bf*)(p + k8);        // regs 0..3
    v8bf hi = *(const v8bf*)(p + 16 + k8);   // regs 4..7
    v16bf r;
#pragma unroll
    for (int i = 0; i < 8; ++i) { r[i] = lo[i]; r[i + 8] = hi[i]; }
    return r;
}

// B-matrix fragment (32x16 bf16, KxN), B[k][n] = W[n][k] (we need inp @ W^T):
//   lane = (khalf<<4)|n ; regs 0-3 hold k = khalf*8 + 0..7, regs 4-7 hold k+16 (W row n)
__device__ __forceinline__ v16bf load_b_frag(const bf16* W, int ntile, int kb, int lane) {
    const int n   = ntile * 16 + (lane & 15);
    const int k16 = (lane >> 4) * 8;
    const bf16* p = W + n * STRIDE + kb + k16;
    v8bf lo = *(const v8bf*)(p);
    v8bf hi = *(const v8bf*)(p + 16);
    v16bf r;
#pragma unroll
    for (int i = 0; i < 8; ++i) { r[i] = lo[i]; r[i + 8] = hi[i]; }
    return r;
}

// Async-copy one 64x512 bf16 activation block (contiguous, 64 KB) from global
// `src` into the padded LDS region at `lds_base_off`, using the CDNA5
// global->LDS async path (GV mode: 64-bit VGPR address + "off").
// 4096 x 16B segments spread over 256 threads.
__device__ __forceinline__ void async_stage(const bf16* src, unsigned lds_base_off, int tid) {
    const unsigned long long sbase = (unsigned long long)src;
#pragma unroll
    for (int it = 0; it < 16; ++it) {
        const int seg    = it * 256 + tid;            // 16-byte segment index
        const int row    = seg >> 6;                  // 64 segs per 1024B row
        const int segrow = seg & 63;
        const unsigned long long gaddr = sbase + (unsigned)(seg * 16);
        const unsigned loff = lds_base_off + (unsigned)(row * (STRIDE * 2) + segrow * 16);
        asm volatile("global_load_async_to_lds_b128 %0, %1, off"
                     :
                     : "v"(loff), "v"(gaddr)
                     : "memory");
    }
}

__global__ void rnn_init_flags(int* __restrict__ flags, int n) {
    int i = blockIdx.x * blockDim.x + threadIdx.x;
    if (i < n) flags[i] = 0;
}

__global__ __launch_bounds__(256, 1)
void rnn_persistent(const float* __restrict__ x,
                    const float* __restrict__ w_ih,
                    const float* __restrict__ b_ih,
                    const float* __restrict__ w_hh,
                    const float* __restrict__ b_hh,
                    float* __restrict__ out,
                    int*   __restrict__ flags,
                    bf16*  __restrict__ ring)   // [RING][NLAYER][BATCH][HDIM] bf16
{
    extern __shared__ bf16 lds[];
    bf16* a_x = lds;                     // [64][STRIDE] staged layer input (bf16)
    bf16* a_h = lds + 1 * 64 * STRIDE;   // [64][STRIDE] staged h(t-1)      (bf16)
    bf16* w_i = lds + 2 * 64 * STRIDE;   // [SLICE][STRIDE] W_ih slice
    bf16* w_h = lds + 3 * 64 * STRIDE;   // [SLICE][STRIDE] W_hh slice

    const int tid   = threadIdx.x;
    const int lane  = tid & 31;
    const int wave  = tid >> 5;          // 8 waves
    const int layer = blockIdx.x / WPL;
    const int slice = blockIdx.x % WPL;
    const int ncol0 = slice * SLICE;

    float* outputs = out;                               // [S,B,H]
    float* states  = out + (size_t)SEQ * BATCH * HDIM;  // [S,L,B,H]

    // ---- one-time: weight slices f32 -> bf16 into LDS ----
    for (int idx = tid; idx < SLICE * (HDIM / 2); idx += 256) {
        const int nl = idx / (HDIM / 2);
        const int k2 = (idx % (HDIM / 2)) * 2;
        const size_t gro = ((size_t)layer * HDIM + ncol0 + nl) * HDIM + k2;
        const float2 wi = *(const float2*)(w_ih + gro);
        const float2 wh = *(const float2*)(w_hh + gro);
        w_i[nl * STRIDE + k2]     = (bf16)wi.x;
        w_i[nl * STRIDE + k2 + 1] = (bf16)wi.y;
        w_h[nl * STRIDE + k2]     = (bf16)wh.x;
        w_h[nl * STRIDE + k2 + 1] = (bf16)wh.y;
    }
    __syncthreads();

    // per-wave tile ownership: 4x4 grid of 16x16 tiles, 2 tiles/wave (same M row)
    const int tia = wave >> 1;            // M tile 0..3
    const int tj0 = (wave & 1) * 2;       // N tile pair {0,1} or {2,3}

    const int nloc0 = tj0 * 16 + (lane & 15);
    const int nloc1 = nloc0 + 16;
    const int ng0 = ncol0 + nloc0;
    const int ng1 = ncol0 + nloc1;
    const float bias0 = b_ih[layer * HDIM + ng0] + b_hh[layer * HDIM + ng0];
    const float bias1 = b_ih[layer * HDIM + ng1] + b_hh[layer * HDIM + ng1];

    for (int t = 0; t < SEQ; ++t) {
        // ---- waits: producer flags + ring back-pressure ----
        if (tid == 0) {
            if (layer > 0) {              // layer-1 finished step t
                const int* f = flags + (t * NLAYER + layer - 1);
                while (__hip_atomic_load(f, __ATOMIC_ACQUIRE, __HIP_MEMORY_SCOPE_AGENT) < WPL)
                    __builtin_amdgcn_s_sleep(1);
            }
            if (t > 0) {                  // our layer finished step t-1 (all slices)
                const int* f = flags + ((t - 1) * NLAYER + layer);
                while (__hip_atomic_load(f, __ATOMIC_ACQUIRE, __HIP_MEMORY_SCOPE_AGENT) < WPL)
                    __builtin_amdgcn_s_sleep(1);
            }
            if (layer < NLAYER - 1 && t >= RING) {
                // before overwriting ring slot t%RING, ensure layer+1 consumed
                // h_layer(t-RING) (implied by it having produced step t-RING)
                const int* f = flags + ((t - RING) * NLAYER + layer + 1);
                while (__hip_atomic_load(f, __ATOMIC_ACQUIRE, __HIP_MEMORY_SCOPE_AGENT) < WPL)
                    __builtin_amdgcn_s_sleep(1);
            }
        }
        __syncthreads();

        // ---- stage activations into LDS ----
        if (layer == 0) {
            // x is f32: convert on the fly
            const float* src_x = x + (size_t)t * BATCH * HDIM;
            for (int idx = tid; idx < BATCH * (HDIM / 2); idx += 256) {
                const int row = idx / (HDIM / 2);
                const int c2  = (idx % (HDIM / 2)) * 2;
                const float2 v = *(const float2*)(src_x + (size_t)row * HDIM + c2);
                a_x[row * STRIDE + c2]     = (bf16)v.x;
                a_x[row * STRIDE + c2 + 1] = (bf16)v.y;
            }
        } else {
            // bf16 ring slot (t % RING, layer-1): async global->LDS copy
            const bf16* src = ring + ((size_t)(t % RING) * NLAYER + (layer - 1)) * BATCH * HDIM;
            async_stage(src, A_X_OFF, tid);
        }
        if (t == 0) {
            for (int idx = tid; idx < BATCH * (HDIM / 2); idx += 256) {
                const int row = idx / (HDIM / 2);
                const int c2  = (idx % (HDIM / 2)) * 2;
                a_h[row * STRIDE + c2]     = (bf16)0.0f;
                a_h[row * STRIDE + c2 + 1] = (bf16)0.0f;
            }
        } else {
            const bf16* src = ring + ((size_t)((t - 1) % RING) * NLAYER + layer) * BATCH * HDIM;
            async_stage(src, A_H_OFF, tid);
        }
        asm volatile("s_wait_asynccnt 0x0" ::: "memory");
        __syncthreads();

        // ---- GEMM: acc = x_t @ W_ih^T + h_prev @ W_hh^T  (K = 512 + 512) ----
        v8f acc0 = {};
        v8f acc1 = {};
#pragma unroll
        for (int kk = 0; kk < HDIM / 32; ++kk) {
            const int kb = kk * 32;
            const v16bf af = load_a_frag(a_x, tia, kb, lane);
            const v16bf b0 = load_b_frag(w_i, tj0,     kb, lane);
            const v16bf b1 = load_b_frag(w_i, tj0 + 1, kb, lane);
            acc0 = __builtin_amdgcn_wmma_f32_16x16x32_bf16(false, af, false, b0,
                                                           (short)0, acc0, false, false);
            acc1 = __builtin_amdgcn_wmma_f32_16x16x32_bf16(false, af, false, b1,
                                                           (short)0, acc1, false, false);
        }
#pragma unroll
        for (int kk = 0; kk < HDIM / 32; ++kk) {
            const int kb = kk * 32;
            const v16bf af = load_a_frag(a_h, tia, kb, lane);
            const v16bf b0 = load_b_frag(w_h, tj0,     kb, lane);
            const v16bf b1 = load_b_frag(w_h, tj0 + 1, kb, lane);
            acc0 = __builtin_amdgcn_wmma_f32_16x16x32_bf16(false, af, false, b0,
                                                           (short)0, acc0, false, false);
            acc1 = __builtin_amdgcn_wmma_f32_16x16x32_bf16(false, af, false, b1,
                                                           (short)0, acc1, false, false);
        }

        // ---- epilogue: bias + tanh; write f32 states/outputs + bf16 ring copy ----
        float* st   = states + (size_t)(t * NLAYER + layer) * BATCH * HDIM;
        bf16*  rg   = ring + ((size_t)(t % RING) * NLAYER + layer) * BATCH * HDIM;
#pragma unroll
        for (int r = 0; r < 8; ++r) {
            const int m = tia * 16 + (lane >> 4) * 8 + r;   // C layout: reg r -> M
            const float v0 = fast_tanh(acc0[r] + bias0);
            const float v1 = fast_tanh(acc1[r] + bias1);
            st[(size_t)m * HDIM + ng0] = v0;
            st[(size_t)m * HDIM + ng1] = v1;
            rg[(size_t)m * HDIM + ng0] = (bf16)v0;
            rg[(size_t)m * HDIM + ng1] = (bf16)v1;
            if (layer == NLAYER - 1) {
                outputs[((size_t)t * BATCH + m) * HDIM + ng0] = v0;
                outputs[((size_t)t * BATCH + m) * HDIM + ng1] = v1;
            }
        }

        __threadfence();      // make h(t) globally visible before signaling
        __syncthreads();
        if (tid == 0) {
            __hip_atomic_fetch_add(&flags[t * NLAYER + layer], 1,
                                   __ATOMIC_RELEASE, __HIP_MEMORY_SCOPE_AGENT);
        }
    }
}

extern "C" void kernel_launch(void* const* d_in, const int* in_sizes, int n_in,
                              void* d_out, int out_size, void* d_ws, size_t ws_size,
                              hipStream_t stream) {
    const float* x    = (const float*)d_in[0];
    const float* w_ih = (const float*)d_in[1];
    const float* b_ih = (const float*)d_in[2];
    const float* w_hh = (const float*)d_in[3];
    const float* b_hh = (const float*)d_in[4];
    float* out   = (float*)d_out;
    int*   flags = (int*)d_ws;
    bf16*  ring  = (bf16*)((char*)d_ws + FLAGS_BYTES);

    const int nflags = SEQ * NLAYER;
    rnn_init_flags<<<(nflags + 255) / 256, 256, 0, stream>>>(flags, nflags);

    rnn_persistent<<<dim3(NLAYER * WPL), dim3(256), LDS_BYTES, stream>>>(
        x, w_ih, b_ih, w_hh, b_hh, out, flags, ring);
}
